// CondDiffusionMambaBlock_10496900072068
// MI455X (gfx1250) — matrix operand order repaired and checked by reading, not compiled, not run on hardware
//
#include <hip/hip_runtime.h>

typedef __attribute__((ext_vector_type(16))) __bf16 v16bf;
typedef __attribute__((ext_vector_type(8)))  __bf16 v8bf;
typedef __attribute__((ext_vector_type(8)))  float  v8f;
typedef __attribute__((ext_vector_type(4)))  int    v4i_t;

#define HW_B      4
#define HW_H      64
#define HW_W      64
#define DMODEL    192
#define DINNER    384
#define DSTATE    16
#define DTRANK    12
#define LSEQ      4096
#define NSEQ      16            // 4 dirs * 4 batch
#define MTOT      (NSEQ * LSEQ) // 65536
#define NTOK      (HW_B * LSEQ) // 16384
#define MODS_N    1152
#define DBLPAD    48            // 44 cols padded to 48
#define SCHUNK    32            // scan LDS chunk (rows of dbl)

// ---------- helpers ----------
__device__ __forceinline__ __bf16 f2bf(float f) {
  union { float f; unsigned u; } v; v.f = f;
  unsigned r = v.u + 0x7FFFu + ((v.u >> 16) & 1u);
  unsigned short h = (unsigned short)(r >> 16);
  __bf16 o; __builtin_memcpy(&o, &h, 2); return o;
}
__device__ __forceinline__ float bf2f(__bf16 b) {
  unsigned short h; __builtin_memcpy(&h, &b, 2);
  union { unsigned u; float f; } v; v.u = ((unsigned)h) << 16; return v.f;
}
__device__ __forceinline__ float silu(float x) { return x / (1.0f + expf(-x)); }
__device__ __forceinline__ float gelu_tanh(float x) {
  float x3 = x * x * x;
  return 0.5f * x * (1.0f + tanhf(0.7978845608028654f * (x + 0.044715f * x3)));
}

// Wait for async global<->LDS transfers (CDNA5 ASYNCcnt domain).
__device__ __forceinline__ void wait_async0() {
#if __has_builtin(__builtin_amdgcn_s_wait_asynccnt)
  __builtin_amdgcn_s_wait_asynccnt(0);
#else
  asm volatile("s_wait_asynccnt 0x0" ::: "memory");
#endif
}

// Cooperative stage of one SCHUNK*DBLPAD f32 chunk (6144 B = 384 x b128) into LDS.
__device__ __forceinline__ void stage_chunk(const float* __restrict__ gsrc,
                                            float* __restrict__ ldst, int tid) {
#if __has_builtin(__builtin_amdgcn_global_load_async_to_lds_b128)
  auto g = (__attribute__((address_space(1))) v4i_t*)gsrc;
  auto l = (__attribute__((address_space(3))) v4i_t*)ldst;
  for (int t = tid; t < 384; t += 256)
    __builtin_amdgcn_global_load_async_to_lds_b128(g + t, l + t, 0, 0);
#else
  const float4* g4 = (const float4*)gsrc;
  float4* l4 = (float4*)ldst;
  for (int t = tid; t < 384; t += 256) l4[t] = g4[t];
#endif
}

// ---------- weight transpose: src [K x N] f32 row-major -> dst [Nout x K] bf16, pad rows >= N with 0
__global__ void wtrans_k(const float* __restrict__ src, __bf16* __restrict__ dst,
                         int K, int N, int Nout) {
  int idx = blockIdx.x * 256 + threadIdx.x;
  if (idx >= Nout * K) return;
  int n = idx / K, k = idx - n * K;
  float v = (n < N) ? src[(size_t)k * N + n] : 0.0f;
  dst[idx] = f2bf(v);
}

// ---------- adaLN mods: mods[b,o] = silu(c[b,:]) @ adaln_w + adaln_b
__global__ void adaln_k(const float* __restrict__ c, const float* __restrict__ w,
                        const float* __restrict__ b, float* __restrict__ mods) {
  int idx = blockIdx.x * 256 + threadIdx.x;
  if (idx >= HW_B * MODS_N) return;
  int bb = idx / MODS_N, o = idx - bb * MODS_N;
  float s = b[o];
  for (int k = 0; k < DMODEL; ++k)
    s += silu(c[bb * DMODEL + k]) * w[(size_t)k * MODS_N + o];
  mods[idx] = s;
}

// ---------- LN1 + modulate + scatter 4 directions (one wave per token, 6 ch/lane)
__global__ __launch_bounds__(256) void prep_k(const float* __restrict__ x,
                                              const float* __restrict__ mods,
                                              float* __restrict__ tokens,
                                              __bf16* __restrict__ xs) {
  int tok = blockIdx.x * 8 + (threadIdx.x >> 5);
  int lane = threadIdx.x & 31;
  int b = tok >> 12, l = tok & (LSEQ - 1);
  int i = l >> 6, j = l & 63;
  float v[6]; float s = 0.f, s2 = 0.f;
#pragma unroll
  for (int q = 0; q < 6; ++q) {
    int cch = lane + q * 32;
    v[q] = x[(((size_t)b * DMODEL + cch) * HW_H + i) * HW_W + j];
    s += v[q]; s2 += v[q] * v[q];
  }
#pragma unroll
  for (int o = 16; o > 0; o >>= 1) { s += __shfl_xor(s, o, 32); s2 += __shfl_xor(s2, o, 32); }
  float mu = s * (1.0f / DMODEL);
  float rs = rsqrtf(s2 * (1.0f / DMODEL) - mu * mu + 1e-6f);
  int l2 = (63 - i) * 64 + (63 - j);
  int l3 = j * 64 + i;
  int l4 = (63 - j) * 64 + (63 - i);
#pragma unroll
  for (int q = 0; q < 6; ++q) {
    int cch = lane + q * 32;
    tokens[(size_t)tok * DMODEL + cch] = v[q];
    float hm = (v[q] - mu) * rs;
    hm = hm * (1.0f + mods[b * MODS_N + DMODEL + cch]) + mods[b * MODS_N + cch];
    __bf16 hb = f2bf(hm);
    xs[(((size_t)(0 * 4 + b)) * LSEQ + l ) * DMODEL + cch] = hb;
    xs[(((size_t)(1 * 4 + b)) * LSEQ + l2) * DMODEL + cch] = hb;
    xs[(((size_t)(2 * 4 + b)) * LSEQ + l3) * DMODEL + cch] = hb;
    xs[(((size_t)(3 * 4 + b)) * LSEQ + l4) * DMODEL + cch] = hb;
  }
}

// ---------- WMMA GEMM with register double-buffering (requires K % 64 == 0)
template<int NT, int ACT, int OBF>
__global__ __launch_bounds__(256) void gemm_bf16_k(const __bf16* __restrict__ A,
                                                   const __bf16* __restrict__ WT,
                                                   const float* __restrict__ bias,
                                                   float* __restrict__ Cf,
                                                   __bf16* __restrict__ Cb,
                                                   int M, int N, int K) {
  const int lane = threadIdx.x & 31;
  const int wave = threadIdx.x >> 5;
  const int half = lane >> 4;   // 0/1 lane-group
  const int lid  = lane & 15;
  const int m0 = blockIdx.y * 128 + wave * 16;
  const int n0 = blockIdx.x * (16 * NT);
  (void)M;

  v8f acc[NT];
#pragma unroll
  for (int t = 0; t < NT; ++t)
#pragma unroll
    for (int r = 0; r < 8; ++r) acc[t][r] = 0.0f;

  const __bf16* arow  = A  + (size_t)(m0 + lid) * K + half * 16;
  const __bf16* bbase = WT + (size_t)(n0 + lid) * K + half * 16;

  auto loadA = [&](int k0, v16bf& a) {
    v8bf alo = *(const v8bf*)(arow + k0);
    v8bf ahi = *(const v8bf*)(arow + k0 + 8);
#pragma unroll
    for (int ii = 0; ii < 8; ++ii) { a[ii] = alo[ii]; a[8 + ii] = ahi[ii]; }
  };
  auto loadB = [&](int k0, int t, v16bf& b) {
    b = *(const v16bf*)(bbase + (size_t)t * 16 * K + k0);
  };

  v16bf a0, a1, b0[NT], b1[NT];
  loadA(0, a0);
#pragma unroll
  for (int t = 0; t < NT; ++t) loadB(0, t, b0[t]);

  for (int k0 = 0; k0 < K; k0 += 64) {
    // prefetch next fragments while current ones are consumed
    loadA(k0 + 32, a1);
#pragma unroll
    for (int t = 0; t < NT; ++t) loadB(k0 + 32, t, b1[t]);
    if (k0 + 128 < K) __builtin_prefetch(arow + k0 + 128, 0, 1);
#pragma unroll
    for (int t = 0; t < NT; ++t)
      acc[t] = __builtin_amdgcn_wmma_f32_16x16x32_bf16(false, a0, false, b0[t],
                                                       (short)0, acc[t], false, false);
    if (k0 + 64 < K) {
      loadA(k0 + 64, a0);
#pragma unroll
      for (int t = 0; t < NT; ++t) loadB(k0 + 64, t, b0[t]);
    }
#pragma unroll
    for (int t = 0; t < NT; ++t)
      acc[t] = __builtin_amdgcn_wmma_f32_16x16x32_bf16(false, a1, false, b1[t],
                                                       (short)0, acc[t], false, false);
  }

#pragma unroll
  for (int t = 0; t < NT; ++t) {
    const int n = n0 + t * 16 + lid;
    const float bv = bias ? bias[n] : 0.0f;
#pragma unroll
    for (int r = 0; r < 8; ++r) {
      const int m = m0 + half * 8 + r;
      float v = acc[t][r] + bv;
      if (ACT == 1) v = gelu_tanh(v);
      if (OBF) Cb[(size_t)m * N + n] = f2bf(v);
      else     Cf[(size_t)m * N + n] = v;
    }
  }
}

// ---------- depthwise causal conv (k=4) + silu on xi half of xz
__global__ void conv_k(const __bf16* __restrict__ xz, const float* __restrict__ cw,
                       const float* __restrict__ cb, __bf16* __restrict__ xc) {
  size_t idx = (size_t)blockIdx.x * 256 + threadIdx.x;
  if (idx >= (size_t)MTOT * DINNER) return;
  int d = (int)(idx % DINNER);
  size_t nl = idx / DINNER;
  int l = (int)(nl & (LSEQ - 1));
  size_t nbase = nl - l;
  float acc = cb[d];
#pragma unroll
  for (int k = 0; k < 4; ++k) {
    int ls = l + k - 3;
    if (ls >= 0) acc += bf2f(xz[(nbase + ls) * (2 * DINNER) + d]) * cw[d * 4 + k];
  }
  xc[idx] = f2bf(silu(acc));
}

// ---------- selective scan with double-buffered async LDS staging of dbl rows
__global__ __launch_bounds__(256) void scan_k(const float* __restrict__ dbl,
                                              const __bf16* __restrict__ xc,
                                              const __bf16* __restrict__ xz,
                                              const float* __restrict__ Wdt,
                                              const float* __restrict__ bdt,
                                              const float* __restrict__ Alog,
                                              const float* __restrict__ Dp,
                                              __bf16* __restrict__ yb) {
  __shared__ __align__(16) float rows[2][SCHUNK * DBLPAD]; // 12 KB
  int n = blockIdx.x;
  int d = blockIdx.y * 16 + (threadIdx.x >> 4);
  int s = threadIdx.x & 15;
  int tid = threadIdx.x;

  float wdt[DTRANK];
#pragma unroll
  for (int r = 0; r < DTRANK; ++r) wdt[r] = Wdt[r * DINNER + d];
  float bd = bdt[d];
  float Aa = -expf(Alog[d * DSTATE + s]);
  float Dd = Dp[d];
  float h = 0.0f;
  size_t base = (size_t)n * LSEQ;
  const float* gsrc = dbl + base * DBLPAD;

  stage_chunk(gsrc, rows[0], tid);
  wait_async0();
  __syncthreads();

  for (int c = 0; c < LSEQ / SCHUNK; ++c) {
    const int cur = c & 1;
    if (c + 1 < LSEQ / SCHUNK)
      stage_chunk(gsrc + (size_t)(c + 1) * SCHUNK * DBLPAD, rows[cur ^ 1], tid);

    const float* rbase = rows[cur];
#pragma unroll 4
    for (int ll = 0; ll < SCHUNK; ++ll) {
      const int l = c * SCHUNK + ll;
      const float* row = rbase + ll * DBLPAD;
      float dtv = bd;
#pragma unroll
      for (int r = 0; r < DTRANK; ++r) dtv += row[r] * wdt[r];
      dtv = (dtv > 20.0f) ? dtv : log1pf(expf(dtv));
      float Bt = row[DTRANK + s];
      float Ct = row[DTRANK + DSTATE + s];
      float xt = bf2f(xc[(base + l) * DINNER + d]);
      h = h * expf(dtv * Aa) + dtv * xt * Bt;
      float p = h * Ct;
      p += __shfl_xor(p, 1, 32); p += __shfl_xor(p, 2, 32);
      p += __shfl_xor(p, 4, 32); p += __shfl_xor(p, 8, 32);
      if (s == 0) {
        float zt = bf2f(xz[(base + l) * (2 * DINNER) + DINNER + d]);
        yb[(base + l) * DINNER + d] = f2bf((p + Dd * xt) * silu(zt));
      }
    }
    wait_async0();
    __syncthreads();
  }
}

// ---------- combine 4 directions + residual1 + LN2 + modulate (wave per token)
__global__ __launch_bounds__(256) void combine_k(const float* __restrict__ ys,
                                                 const float* __restrict__ tokens,
                                                 const float* __restrict__ mods,
                                                 float* __restrict__ tokens2,
                                                 __bf16* __restrict__ mbf) {
  int tok = blockIdx.x * 8 + (threadIdx.x >> 5);
  int lane = threadIdx.x & 31;
  int b = tok >> 12, l = tok & (LSEQ - 1);
  int i = l >> 6, j = l & 63;
  int l2 = (63 - i) * 64 + (63 - j);
  int l3 = j * 64 + i;
  int l4 = (63 - j) * 64 + (63 - i);
  size_t r1 = ((size_t)(0 * 4 + b) * LSEQ + l ) * DMODEL;
  size_t r2 = ((size_t)(1 * 4 + b) * LSEQ + l2) * DMODEL;
  size_t r3 = ((size_t)(2 * 4 + b) * LSEQ + l3) * DMODEL;
  size_t r4 = ((size_t)(3 * 4 + b) * LSEQ + l4) * DMODEL;
  float v[6]; float s = 0.f, s2 = 0.f;
#pragma unroll
  for (int q = 0; q < 6; ++q) {
    int cch = lane + q * 32;
    float ssm = 0.25f * (ys[r1 + cch] + ys[r2 + cch] + ys[r3 + cch] + ys[r4 + cch]);
    float t = tokens[(size_t)tok * DMODEL + cch] + mods[b * MODS_N + 2 * DMODEL + cch] * ssm;
    v[q] = t; s += t; s2 += t * t;
    tokens2[(size_t)tok * DMODEL + cch] = t;
  }
#pragma unroll
  for (int o = 16; o > 0; o >>= 1) { s += __shfl_xor(s, o, 32); s2 += __shfl_xor(s2, o, 32); }
  float mu = s * (1.0f / DMODEL);
  float rs = rsqrtf(s2 * (1.0f / DMODEL) - mu * mu + 1e-6f);
#pragma unroll
  for (int q = 0; q < 6; ++q) {
    int cch = lane + q * 32;
    float m = (v[q] - mu) * rs * (1.0f + mods[b * MODS_N + 4 * DMODEL + cch]) +
              mods[b * MODS_N + 3 * DMODEL + cch];
    mbf[(size_t)tok * DMODEL + cch] = f2bf(m);
  }
}

// ---------- final: residual2 + transpose to NCHW
__global__ void final_k(const float* __restrict__ tokens2, const float* __restrict__ mlp,
                        const float* __restrict__ mods, float* __restrict__ out) {
  size_t idx = (size_t)blockIdx.x * 256 + threadIdx.x;
  if (idx >= (size_t)NTOK * DMODEL) return;
  int cch = (int)(idx % DMODEL);
  size_t tok = idx / DMODEL;
  int b = (int)(tok >> 12); int l = (int)(tok & (LSEQ - 1));
  int i = l >> 6, j = l & 63;
  float t = tokens2[idx] + mods[b * MODS_N + 5 * DMODEL + cch] * mlp[idx];
  out[(((size_t)b * DMODEL + cch) * HW_H + i) * HW_W + j] = t;
}

// =======================================================================
extern "C" void kernel_launch(void* const* d_in, const int* in_sizes, int n_in,
                              void* d_out, int out_size, void* d_ws, size_t ws_size,
                              hipStream_t stream) {
  (void)in_sizes; (void)n_in; (void)out_size; (void)ws_size;
  const float* x_in   = (const float*)d_in[0];
  const float* c_in   = (const float*)d_in[1];
  const float* adaw   = (const float*)d_in[2];
  const float* adab   = (const float*)d_in[3];
  const float* W_in   = (const float*)d_in[4];
  const float* b_in   = (const float*)d_in[5];
  const float* conv_w = (const float*)d_in[6];
  const float* conv_b = (const float*)d_in[7];
  const float* W_x    = (const float*)d_in[8];
  const float* W_dt   = (const float*)d_in[9];
  const float* b_dt   = (const float*)d_in[10];
  const float* A_log  = (const float*)d_in[11];
  const float* Dvec   = (const float*)d_in[12];
  const float* W_out  = (const float*)d_in[13];
  const float* b_out  = (const float*)d_in[14];
  const float* mlpw1  = (const float*)d_in[15];
  const float* mlpb1  = (const float*)d_in[16];
  const float* mlpw2  = (const float*)d_in[17];
  const float* mlpb2  = (const float*)d_in[18];
  float* out = (float*)d_out;

  char* W = (char*)d_ws;
  size_t off = 0;
  auto take = [&](size_t bytes) { size_t o = off; off += (bytes + 255) & ~(size_t)255; return o; };

  float*  mods    = (float*) (W + take((size_t)HW_B * MODS_N * 4));
  float*  tokens  = (float*) (W + take((size_t)NTOK * DMODEL * 4));
  char*   xsReg   =           W + take((size_t)MTOT * DMODEL * 2);      // xs_bf  -> later h1_bf
  char*   xzReg   =           W + take((size_t)MTOT * 2 * DINNER * 2);  // xz_bf  -> later tokens2
  __bf16* xc_bf   = (__bf16*)(W + take((size_t)MTOT * DINNER * 2));
  char*   dblReg  =           W + take((size_t)MTOT * DBLPAD * 4);      // dbl f32 -> later m_bf
  char*   ybReg   =           W + take((size_t)MTOT * DINNER * 2);      // y_bf -> later mlp f32
  float*  ys      = (float*) (W + take((size_t)MTOT * DMODEL * 4));
  __bf16* winT    = (__bf16*)(W + take((size_t)768 * 192 * 2));
  __bf16* wxT     = (__bf16*)(W + take((size_t)DBLPAD * DINNER * 2));
  __bf16* woutT   = (__bf16*)(W + take((size_t)192 * 384 * 2));
  __bf16* m1T     = (__bf16*)(W + take((size_t)768 * 192 * 2));
  __bf16* m2T     = (__bf16*)(W + take((size_t)192 * 768 * 2));

  __bf16* xs_bf   = (__bf16*)xsReg;
  __bf16* h1_bf   = (__bf16*)xsReg;     // alias (xs dead after W_in GEMM)
  __bf16* xz_bf   = (__bf16*)xzReg;
  float*  tokens2 = (float*) xzReg;     // alias (xz dead after scan)
  float*  dbl     = (float*) dblReg;
  __bf16* m_bf    = (__bf16*)dblReg;    // alias (dbl dead after scan)
  __bf16* y_bf    = (__bf16*)ybReg;
  float*  mlp     = (float*) ybReg;     // alias (y_bf dead after W_out GEMM)

  // --- weight transpose/convert to bf16 [N x K]
  wtrans_k<<<(768 * 192 + 255) / 256, 256, 0, stream>>>(W_in,  winT, 192, 768, 768);
  wtrans_k<<<(DBLPAD * 384 + 255) / 256, 256, 0, stream>>>(W_x, wxT,  384, 44, DBLPAD);
  wtrans_k<<<(192 * 384 + 255) / 256, 256, 0, stream>>>(W_out, woutT, 384, 192, 192);
  wtrans_k<<<(768 * 192 + 255) / 256, 256, 0, stream>>>(mlpw1, m1T, 192, 768, 768);
  wtrans_k<<<(192 * 768 + 255) / 256, 256, 0, stream>>>(mlpw2, m2T, 768, 192, 192);

  // --- adaLN mods
  adaln_k<<<(HW_B * MODS_N + 255) / 256, 256, 0, stream>>>(c_in, adaw, adab, mods);

  // --- LN1 + modulate + scatter 4 scan directions
  prep_k<<<NTOK / 8, 256, 0, stream>>>(x_in, mods, tokens, xs_bf);

  // --- xz = xs @ W_in + b_in            [65536 x 768], K=192
  gemm_bf16_k<4, 0, 1><<<dim3(768 / 64, MTOT / 128), 256, 0, stream>>>(
      xs_bf, winT, b_in, nullptr, xz_bf, MTOT, 768, 192);

  // --- depthwise conv + silu -> xc
  conv_k<<<(unsigned)(((size_t)MTOT * DINNER) / 256), 256, 0, stream>>>(xz_bf, conv_w, conv_b, xc_bf);

  // --- dbl = xc @ W_x (padded to 48)    [65536 x 48], K=384
  gemm_bf16_k<3, 0, 0><<<dim3(1, MTOT / 128), 256, 0, stream>>>(
      xc_bf, wxT, nullptr, dbl, nullptr, MTOT, DBLPAD, 384);

  // --- selective scan (fused dt-proj + softplus + D*x + silu(z) gate)
  scan_k<<<dim3(NSEQ, DINNER / 16), 256, 0, stream>>>(
      dbl, xc_bf, xz_bf, W_dt, b_dt, A_log, Dvec, y_bf);

  // --- ys = y @ W_out + b_out           [65536 x 192], K=384
  gemm_bf16_k<4, 0, 0><<<dim3(192 / 64, MTOT / 128), 256, 0, stream>>>(
      y_bf, woutT, b_out, ys, nullptr, MTOT, 192, 384);

  // --- combine 4 dirs, residual1, LN2 + modulate
  combine_k<<<NTOK / 8, 256, 0, stream>>>(ys, tokens, mods, tokens2, m_bf);

  // --- h1 = gelu(m @ mlp_w1 + b1)       [16384 x 768], K=192
  gemm_bf16_k<4, 1, 1><<<dim3(768 / 64, NTOK / 128), 256, 0, stream>>>(
      m_bf, m1T, mlpb1, nullptr, h1_bf, NTOK, 768, 192);

  // --- mlp = h1 @ mlp_w2 + b2           [16384 x 192], K=768
  gemm_bf16_k<4, 0, 0><<<dim3(192 / 64, NTOK / 128), 256, 0, stream>>>(
      h1_bf, m2T, mlpb2, mlp, nullptr, NTOK, 192, 768);

  // --- residual2 + NCHW transpose
  final_k<<<(unsigned)(((size_t)NTOK * DMODEL) / 256), 256, 0, stream>>>(tokens2, mlp, mods, out);
}
